// DeepGSEA_45191645888600
// MI455X (gfx1250) — matrix operand order repaired and checked by reading, not compiled, not run on hardware
//
#include <hip/hip_runtime.h>
#include <math.h>

typedef __attribute__((ext_vector_type(16))) _Float16 v16h;
typedef __attribute__((ext_vector_type(8)))  _Float16 v8h;
typedef __attribute__((ext_vector_type(2)))  _Float16 v2h;
typedef __attribute__((ext_vector_type(8)))  float    v8f;

#define Bn   2048
#define Gn   2000
#define Cn   50
#define Hn   256
#define ZDn  32
#define Kn   10
#define Pn   4

#define BT   64          // batch rows per block
#define KT   32          // WMMA K step (f16)
#define ASTR 40          // A-tile LDS stride (halfs, %8==0 for aligned b128)
#define BSTR 40          // transposed B-tile LDS stride (halfs, %8==0)
#define HSTR 264         // Hs LDS stride (halfs, %8==0)
#define LREL 0.01f       // jax.nn.leaky_relu default slope

// ---- static LDS carving ----
#define ALS_OFF 0
#define BT_OFF  (BT * ASTR * 2)               // 5120
#define ZS_OFF  (BT_OFF + 32 * BSTR * 2)      // 7680: overlays Bt rows>=32 (live only in stages 3-4)
#define HS_OFF  (BT_OFF + 256 * BSTR * 2)     // 25600
#define SMEM_BYTES (HS_OFF + BT * HSTR * 2)   // 59392 (< 64KB)

// A-matrix 16x32 f16 fragment (row-major tile): lane -> M = lane&15;
// K base = (lane>=16?16:0); 16 contiguous halfs. Two aligned b128 LDS loads.
__device__ __forceinline__ v16h load_a_frag(const _Float16* base, int stride, int lane) {
  const _Float16* p = base + (lane & 15) * stride + ((lane >> 4) << 4);
  v8h lo = *(const v8h*)(p);
  v8h hi = *(const v8h*)(p + 8);
  return __builtin_shufflevector(lo, hi, 0, 1, 2, 3, 4, 5, 6, 7, 8, 9, 10, 11, 12, 13, 14, 15);
}

// B-matrix 32x16 f16 fragment from TRANSPOSED tile Bt[n][k]: lane -> N = lane&15,
// K base = (lane>=16?16:0); 16 contiguous halfs -> two aligned b128 LDS loads.
__device__ __forceinline__ v16h load_b_frag(const _Float16* base, int stride, int lane) {
  const _Float16* p = base + (lane & 15) * stride + ((lane >> 4) << 4);
  v8h lo = *(const v8h*)(p);
  v8h hi = *(const v8h*)(p + 8);
  return __builtin_shufflevector(lo, hi, 0, 1, 2, 3, 4, 5, 6, 7, 8, 9, 10, 11, 12, 13, 14, 15);
}

__device__ __forceinline__ v8f wmma_f16(v16h a, v16h b, v8f c) {
  return __builtin_amdgcn_wmma_f32_16x16x32_f16(false, a, false, b, (short)0, c, false, false);
}

__global__ __launch_bounds__(256)
void deepgsea_fused(const float* __restrict__ X,   const float* __restrict__ Mm,
                    const float* __restrict__ W1,  const float* __restrict__ b1,
                    const float* __restrict__ W2,  const float* __restrict__ b2,
                    const float* __restrict__ cw,  const float* __restrict__ cb,
                    const float* __restrict__ proto, const float* __restrict__ logvar,
                    const float* __restrict__ cbias, float* __restrict__ c_logits) {
  __shared__ __align__(16) unsigned char smem[SMEM_BYTES];
  _Float16* Als = (_Float16*)(smem + ALS_OFF);   // [BT][ASTR] masked-X tile
  _Float16* Bt  = (_Float16*)(smem + BT_OFF);    // [256][BSTR] transposed B tile: Bt[n][k]
  _Float16* Hs  = (_Float16*)(smem + HS_OFF);    // [BT][HSTR] h1 then h2 (in place)
  float*    Zs  = (float*)   (smem + ZS_OFF);    // [BT][ZDn]  (overlays Bt rows>=32)

  const int tid  = threadIdx.x;
  const int lane = tid & 31;
  const int w    = tid >> 5;          // wave 0..7
  const int b0   = blockIdx.x * BT;
  const int c    = blockIdx.y;
  const int n0   = w * 32;            // wave's 32-wide N strip (stages 1-2)

  const v8f zero8 = {0.f, 0.f, 0.f, 0.f, 0.f, 0.f, 0.f, 0.f};
  v8f acc[4][2];
#pragma unroll
  for (int i = 0; i < 4; ++i) { acc[i][0] = zero8; acc[i][1] = zero8; }

  // ---------------- Stage 1: h1 = leaky((X*M[c]) @ W1 + b1) ----------------
  const int KI = (Gn + KT - 1) / KT;   // 63 (last tile zero-padded; pairs never straddle: Gn even)
  for (int it = 0; it < KI; ++it) {
    const int k0 = it * KT;
    // A tile: 64x32 masked X, float2 loads, packed v2h stores
#pragma unroll
    for (int j = 0; j < 4; ++j) {
      const int pi = tid + j * 256;        // 0..1023 pairs
      const int r  = pi >> 4;              // row 0..63
      const int c2 = (pi & 15) << 1;       // k pair base 0..30
      const int g  = k0 + c2;
      float2 xv = make_float2(0.f, 0.f), mv = make_float2(0.f, 0.f);
      if (g + 1 < Gn) {
        xv = *(const float2*)&X [(size_t)(b0 + r) * Gn + g];
        mv = *(const float2*)&Mm[(size_t)c * Gn + g];
      }
      v2h pk = { (_Float16)(xv.x * mv.x), (_Float16)(xv.y * mv.y) };
      *(v2h*)&Als[r * ASTR + c2] = pk;
    }
    // B tile: W1[k0+k][n] -> Bt[n][k], two rows packed per b32 store
#pragma unroll
    for (int j = 0; j < 16; ++j) {
      const int pi = tid + j * 256;        // 0..4095 pairs
      const int n  = pi & 255;
      const int rp = pi >> 8;              // 0..15
      const int g0 = k0 + 2 * rp;
      const float w0 = (g0     < Gn) ? W1[(size_t)g0 * Hn + n]       : 0.f;
      const float w1 = (g0 + 1 < Gn) ? W1[(size_t)(g0 + 1) * Hn + n] : 0.f;
      v2h pk = { (_Float16)w0, (_Float16)w1 };
      *(v2h*)&Bt[n * BSTR + 2 * rp] = pk;
    }
    if (it + 1 < KI) __builtin_prefetch(&W1[(size_t)(k0 + KT) * Hn], 0, 1);
    __syncthreads();
    v16h bf0 = load_b_frag(&Bt[(n0     ) * BSTR], BSTR, lane);
    v16h bf1 = load_b_frag(&Bt[(n0 + 16) * BSTR], BSTR, lane);
#pragma unroll
    for (int mt = 0; mt < 4; ++mt) {
      v16h af = load_a_frag(&Als[mt * 16 * ASTR], ASTR, lane);
      acc[mt][0] = wmma_f16(af, bf0, acc[mt][0]);
      acc[mt][1] = wmma_f16(af, bf1, acc[mt][1]);
    }
    __syncthreads();
  }
  // bias + leaky-relu -> Hs
#pragma unroll
  for (int mt = 0; mt < 4; ++mt)
#pragma unroll
    for (int nt = 0; nt < 2; ++nt) {
      const int col = n0 + nt * 16 + (lane & 15);
      const float bv = b1[col];
#pragma unroll
      for (int r = 0; r < 8; ++r) {
        const int row = mt * 16 + r + ((lane >> 4) << 3);
        float v = acc[mt][nt][r] + bv;
        v = v > 0.f ? v : LREL * v;
        Hs[row * HSTR + col] = (_Float16)v;
      }
    }
#pragma unroll
  for (int i = 0; i < 4; ++i) { acc[i][0] = zero8; acc[i][1] = zero8; }

  // ---------------- Stage 2: h2 = leaky(h1 @ W2 + b2), in place ----------------
  for (int it = 0; it < Hn / KT; ++it) {   // 8 iterations
    const int k0 = it * KT;
#pragma unroll
    for (int j = 0; j < 16; ++j) {
      const int pi = tid + j * 256;
      const int n  = pi & 255;
      const int rp = pi >> 8;
      const int g0 = k0 + 2 * rp;
      v2h pk = { (_Float16)W2[(size_t)g0 * Hn + n], (_Float16)W2[(size_t)(g0 + 1) * Hn + n] };
      *(v2h*)&Bt[n * BSTR + 2 * rp] = pk;
    }
    __syncthreads();   // also orders Hs stores vs. A-frag reads
    v16h bf0 = load_b_frag(&Bt[(n0     ) * BSTR], BSTR, lane);
    v16h bf1 = load_b_frag(&Bt[(n0 + 16) * BSTR], BSTR, lane);
#pragma unroll
    for (int mt = 0; mt < 4; ++mt) {
      v16h af = load_a_frag(&Hs[mt * 16 * HSTR + k0], HSTR, lane);
      acc[mt][0] = wmma_f16(af, bf0, acc[mt][0]);
      acc[mt][1] = wmma_f16(af, bf1, acc[mt][1]);
    }
    __syncthreads();
  }
  // bias + leaky -> Hs in place (all h1 reads completed at last barrier)
#pragma unroll
  for (int mt = 0; mt < 4; ++mt)
#pragma unroll
    for (int nt = 0; nt < 2; ++nt) {
      const int col = n0 + nt * 16 + (lane & 15);
      const float bv = b2[col];
#pragma unroll
      for (int r = 0; r < 8; ++r) {
        const int row = mt * 16 + r + ((lane >> 4) << 3);
        float v = acc[mt][nt][r] + bv;
        v = v > 0.f ? v : LREL * v;
        Hs[row * HSTR + col] = (_Float16)v;
      }
    }

  // ---------------- Stage 3: Z = h2 @ cw[c]^T + cb[c] (64x32) ----------------
  v8f acc3 = zero8;
  const int mt3 = w & 3;
  const int nt3 = w >> 2;
  for (int it = 0; it < Hn / KT; ++it) {
    const int k0 = it * KT;
    // B tile 32x32: Bt[n][k] = cw[c, n, k0+k] -- coalesced float2 reads
#pragma unroll
    for (int j = 0; j < 2; ++j) {
      const int pi = tid + j * 256;        // 0..511 pairs
      const int n  = pi >> 4;              // 0..31
      const int c2 = (pi & 15) << 1;       // 0..30
      const float2 wv = *(const float2*)&cw[((size_t)c * ZDn + n) * Hn + k0 + c2];
      v2h pk = { (_Float16)wv.x, (_Float16)wv.y };
      *(v2h*)&Bt[n * BSTR + c2] = pk;
    }
    __syncthreads();
    v16h bf = load_b_frag(&Bt[nt3 * 16 * BSTR], BSTR, lane);
    v16h af = load_a_frag(&Hs[mt3 * 16 * HSTR + k0], HSTR, lane);
    acc3 = wmma_f16(af, bf, acc3);
    __syncthreads();
  }
  {
    const int col = nt3 * 16 + (lane & 15);
    const float cbv = cb[c * ZDn + col];
#pragma unroll
    for (int r = 0; r < 8; ++r) {
      const int row = mt3 * 16 + r + ((lane >> 4) << 3);
      Zs[row * ZDn + col] = acc3[r] + cbv;   // Zs overlays Bt rows>=32: no clash with rows 0..31
    }
  }
  __syncthreads();

  // ---------------- Stage 4: prototype distances -> c_logits ----------------
  for (int idx = tid; idx < BT * Kn; idx += 256) {
    const int row = idx / Kn;
    const int k   = idx - row * Kn;
    float best = 3.4e38f;
#pragma unroll
    for (int p = 0; p < Pn; ++p) {
      const float inv = 0.5f / expf(logvar[(c * Kn + k) * Pn + p]);
      const float* pp = &proto[(size_t)((c * Kn + k) * Pn + p) * ZDn];
      float d = 0.f;
#pragma unroll
      for (int z = 0; z < ZDn; ++z) {
        const float df = Zs[row * ZDn + z] - pp[z];
        d += df * df;
      }
      const float s = d * inv;
      best = s < best ? s : best;
    }
    c_logits[((size_t)(b0 + row) * Cn + c) * Kn + k] = -best + cbias[c * Kn + k];
  }
}

// logits[b,k] = log(sum_c exp(c_logits[b,c,k]) * |imp[c]| + 1e-16) + bias[k]
__global__ __launch_bounds__(256)
void final_logits(const float* __restrict__ c_logits, const float* __restrict__ imp,
                  const float* __restrict__ bias, float* __restrict__ logits) {
  const int t = blockIdx.x * blockDim.x + threadIdx.x;
  if (t >= Bn * Kn) return;
  const int b = t / Kn;
  const int k = t - b * Kn;
  float s = 0.f;
#pragma unroll 5
  for (int c = 0; c < Cn; ++c)
    s += expf(c_logits[((size_t)b * Cn + c) * Kn + k]) * fabsf(imp[c]);
  logits[t] = logf(s + 1e-16f) + bias[k];
}

extern "C" void kernel_launch(void* const* d_in, const int* in_sizes, int n_in,
                              void* d_out, int out_size, void* d_ws, size_t ws_size,
                              hipStream_t stream) {
  (void)in_sizes; (void)n_in; (void)out_size; (void)d_ws; (void)ws_size;
  const float* X      = (const float*)d_in[0];
  const float* Mm     = (const float*)d_in[1];
  const float* W1     = (const float*)d_in[2];
  const float* b1     = (const float*)d_in[3];
  const float* W2     = (const float*)d_in[4];
  const float* b2     = (const float*)d_in[5];
  const float* cw     = (const float*)d_in[6];
  const float* cb     = (const float*)d_in[7];
  const float* proto  = (const float*)d_in[8];
  const float* logvar = (const float*)d_in[9];
  const float* cbias  = (const float*)d_in[10];
  const float* bias   = (const float*)d_in[11];
  const float* imp    = (const float*)d_in[12];

  float* out      = (float*)d_out;
  float* logits   = out;               // [B, K]
  float* c_logits = out + Bn * Kn;     // [B, C, K]

  dim3 grid(Bn / BT, Cn);
  deepgsea_fused<<<grid, 256, 0, stream>>>(X, Mm, W1, b1, W2, b2, cw, cb,
                                           proto, logvar, cbias, c_logits);

  const int tot = Bn * Kn;
  final_logits<<<(tot + 255) / 256, 256, 0, stream>>>(c_logits, imp, bias, logits);
}
